// ObjectSelector_23656679867548
// MI455X (gfx1250) — matrix operand order repaired and checked by reading, not compiled, not run on hardware
//
#include <hip/hip_runtime.h>
#include <hip/hip_bf16.h>

// ---------------------------------------------------------------------------
// ObjectSelector fused pipeline for MI455X (gfx1250, wave32, WMMA)
//
//  objects(262144x128) -relu-> x0(x256) -relu-> x1(x256) -> kv(x512)
//  k half consumed in-register for logits = (q[seg] . k)/16
//  v half staged in LDS then coalesced-stored bf16 to workspace;
//  pass 2 does segment softmax + sum(w*v)
// ---------------------------------------------------------------------------

#define N_TOTAL 262144
#define BATCH   4096
#define HDIM    256
#define OSIZE   128
#define CSIZE   256

#define M_TILE  64      // rows per block (4 waves x 16 rows)
#define XS      264     // padded LDS row stride (elements, multiple of 8 -> 16B aligned)

typedef __attribute__((ext_vector_type(16))) __bf16 v16bf;
typedef __attribute__((ext_vector_type(8)))  float  v8f;

union FragBF {
    v16bf v;
    uint4 q[2];
    unsigned short us[16];
};

// native bf16 convert (v_cvt_*_bf16 on gfx1250) instead of manual RNE bit math
__device__ __forceinline__ unsigned short f32_to_bf16_hw(float f) {
    __bf16 h = (__bf16)f;
    return __builtin_bit_cast(unsigned short, h);
}
__device__ __forceinline__ float bf16_to_f32(unsigned short h) {
    return __uint_as_float(((unsigned int)h) << 16);
}

// ---------------------------------------------------------------------------
// Kernel 1: pre-swizzle a f32 weight matrix (Ktot x Ntot, row major) into
// WMMA B-fragment order: frag f = nt*(Ktot/32)+kt, lane-major, 16 bf16/lane.
// B order used here (contiguous k, matching the A loader): lane L elem e -> K=(L>>4)*16+e, N=L&15.
// ---------------------------------------------------------------------------
__global__ void pack_weight_frags(const float* __restrict__ W,
                                  unsigned short* __restrict__ out,
                                  int Ktot, int Ntot) {
    int t    = blockIdx.x * blockDim.x + threadIdx.x;
    int lane = t & 31;
    int f    = t >> 5;
    int kTiles = Ktot >> 5;
    int nTiles = Ntot >> 4;
    if (f >= kTiles * nTiles) return;
    int kt = f % kTiles;
    int nt = f / kTiles;
    int n     = nt * 16 + (lane & 15);
    int kbase = kt * 32 + ((lane >> 4) << 4);
    unsigned short* dst = out + (size_t)f * 512 + lane * 16;
#pragma unroll
    for (int e = 0; e < 16; ++e)
        dst[e] = f32_to_bf16_hw(W[(size_t)(kbase + e) * Ntot + n]);
}

// ---------------------------------------------------------------------------
// Kernel 2: q = context @ Wq + bq   (4096 x 256, tiny: plain FMA)
// ---------------------------------------------------------------------------
__global__ __launch_bounds__(256)
void compute_q(const float* __restrict__ context,
               const float* __restrict__ Wq,
               const float* __restrict__ bq,
               float* __restrict__ qout) {
    __shared__ float ctx[CSIZE];
    int b = blockIdx.x, t = threadIdx.x;
    ctx[t] = context[(size_t)b * CSIZE + t];
    __syncthreads();
    float acc = bq[t];
#pragma unroll 8
    for (int k = 0; k < CSIZE; ++k)
        acc = fmaf(ctx[k], Wq[(size_t)k * HDIM + t], acc);
    qout[(size_t)b * HDIM + t] = acc;
}

// ---------------------------------------------------------------------------
// A-fragment loader from a bf16 LDS tile (row-major, stride XS).
// 16-bit A layout (contiguous k order, matching packed B): lane L, elem e -> M = L&15,
// K = (L>>4)*16 + e  => one contiguous 32B run per lane.
// ---------------------------------------------------------------------------
__device__ __forceinline__ v16bf load_a_frag(const unsigned short* tile,
                                             int lane, int rowBase16, int kt) {
    FragBF a;
    int r  = rowBase16 + (lane & 15);
    int kb = kt * 32 + ((lane >> 4) << 4);
    const unsigned short* p = tile + (size_t)r * XS + kb;
    a.q[0] = *(const uint4*)(p);        // K = kb .. kb+7
    a.q[1] = *(const uint4*)(p + 8);    // K = kb+8 .. kb+15
    return a.v;
}

// C fragment (f32, 8 slots/lane): slot s, lane L -> M = s + (L>>4)*8, N = L&15

// Per-wave 16x256 GEMM layer with bias+relu, bf16 out to LDS.
__device__ __forceinline__ void wave_layer_relu(const unsigned short* tin, int kTiles,
                                                const unsigned short* __restrict__ wb,
                                                const float* __restrict__ bias,
                                                unsigned short* tout,
                                                int lane, int wave) {
    const int col16 = lane & 15;
    const int mhi   = (lane >> 4) << 3;
#pragma unroll 4
    for (int nt = 0; nt < 16; ++nt) {
        v8f acc = {};
        const unsigned short* wp = wb + ((size_t)nt * kTiles) * 512 + lane * 16;
        for (int kt = 0; kt < kTiles; ++kt) {
            v16bf a = load_a_frag(tin, lane, wave * 16, kt);
            FragBF b;
            b.q[0] = *(const uint4*)(wp);
            b.q[1] = *(const uint4*)(wp + 8);
            __builtin_prefetch(wp + 512, 0, 1);   // next fragment -> global_prefetch
            wp += 512;
            acc = __builtin_amdgcn_wmma_f32_16x16x32_bf16(
                false, a, false, b.v, (short)0, acc, false, false);
        }
        float bv  = bias[nt * 16 + col16];
        int   col = nt * 16 + col16;
#pragma unroll
        for (int s = 0; s < 8; ++s) {
            float x = fmaxf(acc[s] + bv, 0.f);    // relu -> v_max_num_f32
            int row = wave * 16 + s + mhi;
            tout[(size_t)row * XS + col] = f32_to_bf16_hw(x);
        }
    }
}

// ---------------------------------------------------------------------------
// Kernel 3: fused MLP chain + kv + logits.  One block = 64 rows (4 waves).
// ---------------------------------------------------------------------------
__global__ __launch_bounds__(128)
void fused_mlp_attn(const float* __restrict__ objects,
                    const int* __restrict__ seg_ids,
                    const unsigned short* __restrict__ w0b,
                    const unsigned short* __restrict__ w1b,
                    const unsigned short* __restrict__ wkvb,
                    const float* __restrict__ b0,
                    const float* __restrict__ b1,
                    const float* __restrict__ bkv,
                    const float* __restrict__ qmat,
                    unsigned short* __restrict__ v_out,
                    float* __restrict__ logits) {
    extern __shared__ uint4 smem4[];
    unsigned short* bufA = (unsigned short*)smem4;          // objects, then x1
    unsigned short* bufB = bufA + (size_t)M_TILE * XS;      // x0, then v tile

    const int tid  = threadIdx.x;
    const int lane = tid & 31;
    const int wave = tid >> 5;
    const int row0 = blockIdx.x * M_TILE;
    const int col16 = lane & 15;
    const int mhi   = (lane >> 4) << 3;

    // Stage 0: objects (64 x 128 f32) -> bf16 LDS tile.
    // float4 coalesced loads, packed 4x bf16 (8B) LDS stores.
    for (int i = tid; i < (M_TILE * OSIZE) / 4; i += blockDim.x) {
        int e = i * 4;
        int r = e >> 7;              // /128
        int c = e & (OSIZE - 1);
        float4 f = *(const float4*)(objects + (size_t)(row0 + r) * OSIZE + c);
        ushort4 h;
        h.x = f32_to_bf16_hw(f.x);
        h.y = f32_to_bf16_hw(f.y);
        h.z = f32_to_bf16_hw(f.z);
        h.w = f32_to_bf16_hw(f.w);
        *(ushort4*)(bufA + (size_t)r * XS + c) = h;
    }
    __syncthreads();

    // Layer 0: x0 = relu(obj @ W0 + b0)   K=128 (4 k-steps)
    wave_layer_relu(bufA, OSIZE / 32, w0b, b0, bufB, lane, wave);
    __syncthreads();

    // Layer 1: x1 = relu(x0 @ W1 + b1)    K=256 (8 k-steps)
    wave_layer_relu(bufB, HDIM / 32, w1b, b1, bufA, lane, wave);
    __syncthreads();

    // KV layer: kv = x1 @ Wkv + bkv  (N=512: nt<16 -> k half, nt>=16 -> v half)
    int segr[8];
#pragma unroll
    for (int s = 0; s < 8; ++s)
        segr[s] = seg_ids[row0 + wave * 16 + s + mhi];

    float part[8] = {0.f, 0.f, 0.f, 0.f, 0.f, 0.f, 0.f, 0.f};

    for (int nt = 0; nt < 32; ++nt) {
        v8f acc = {};
        const unsigned short* wp = wkvb + ((size_t)nt * 8) * 512 + lane * 16;
        for (int kt = 0; kt < 8; ++kt) {
            v16bf a = load_a_frag(bufA, lane, wave * 16, kt);
            FragBF b;
            b.q[0] = *(const uint4*)(wp);
            b.q[1] = *(const uint4*)(wp + 8);
            __builtin_prefetch(wp + 512, 0, 1);
            wp += 512;
            acc = __builtin_amdgcn_wmma_f32_16x16x32_bf16(
                false, a, false, b.v, (short)0, acc, false, false);
        }
        int   col = nt * 16 + col16;
        float bv  = bkv[col];
        if (nt < 16) {            // k half: consume in-register for logits
#pragma unroll
            for (int s = 0; s < 8; ++s) {
                float kval = acc[s] + bv;
                part[s] = fmaf(qmat[(size_t)segr[s] * HDIM + col], kval, part[s]);
            }
        } else {                  // v half: stage bf16 into LDS (bufB is free)
            int vcol = col - HDIM;
#pragma unroll
            for (int s = 0; s < 8; ++s) {
                float vval = acc[s] + bv;
                int row = wave * 16 + s + mhi;
                bufB[(size_t)row * XS + vcol] = f32_to_bf16_hw(vval);
            }
        }
    }

    // Reduce logit partials across the 16 lanes of each half-wave
#pragma unroll
    for (int m = 1; m <= 8; m <<= 1) {
#pragma unroll
        for (int s = 0; s < 8; ++s)
            part[s] += __shfl_xor(part[s], m, 32);
    }
    if (col16 == 0) {
#pragma unroll
        for (int s = 0; s < 8; ++s) {
            int row = row0 + wave * 16 + s + mhi;
            logits[row] = part[s] * 0.0625f;   // 1/sqrt(256)
        }
    }
    __syncthreads();

    // Coalesced v tile writeback: 16B ds_load_b128 -> global_store_b128
    for (int i = tid; i < (M_TILE * HDIM) / 8; i += blockDim.x) {
        int e = i * 8;
        int r = e >> 8;              // /256
        int c = e & (HDIM - 1);
        uint4 pack = *(const uint4*)(bufB + (size_t)r * XS + c);
        *(uint4*)(v_out + (size_t)(row0 + r) * HDIM + c) = pack;
    }
}

// ---------------------------------------------------------------------------
// Kernel 4: per-segment softmax + weighted v-sum.  One block per segment;
// segment_ids are sorted, so binary-search the range.
// ---------------------------------------------------------------------------
__device__ __forceinline__ int lower_bound_dev(const int* a, int n, int v) {
    int lo = 0, hi = n;
    while (lo < hi) {
        int mid = (lo + hi) >> 1;
        if (a[mid] < v) lo = mid + 1; else hi = mid;
    }
    return lo;
}

__global__ __launch_bounds__(256)
void segment_softmax_embed(const float* __restrict__ logits,
                           const int* __restrict__ seg_ids,
                           const unsigned short* __restrict__ v_in,
                           float* __restrict__ emb_out,
                           float* __restrict__ w_out) {
    __shared__ float red[256];
    __shared__ int   sse[2];
    __shared__ float stat[2];

    const int sid = blockIdx.x;
    const int tid = threadIdx.x;

    if (tid == 0) sse[0] = lower_bound_dev(seg_ids, N_TOTAL, sid);
    if (tid == 1) sse[1] = lower_bound_dev(seg_ids, N_TOTAL, sid + 1);
    __syncthreads();
    const int start = sse[0], end = sse[1];

    // segment max
    float mx = -3.4e38f;
    for (int i = start + tid; i < end; i += 256) mx = fmaxf(mx, logits[i]);
    red[tid] = mx; __syncthreads();
    for (int o = 128; o > 0; o >>= 1) {
        if (tid < o) red[tid] = fmaxf(red[tid], red[tid + o]);
        __syncthreads();
    }
    if (tid == 0) stat[0] = red[0];
    __syncthreads();
    const float m = stat[0];

    // exp sum
    float sum = 0.f;
    for (int i = start + tid; i < end; i += 256) sum += __expf(logits[i] - m);
    red[tid] = sum; __syncthreads();
    for (int o = 128; o > 0; o >>= 1) {
        if (tid < o) red[tid] += red[tid + o];
        __syncthreads();
    }
    if (tid == 0) stat[1] = 1.0f / fmaxf(red[0], 1e-9f);
    __syncthreads();
    const float inv = stat[1];

    // per-object weights
    for (int i = start + tid; i < end; i += 256)
        w_out[i] = __expf(logits[i] - m) * inv;

    // embedding: tid == feature dim (HDIM == 256 == blockDim.x)
    float acc = 0.f;
    for (int i = start; i < end; ++i) {
        float w = __expf(logits[i] - m) * inv;   // scalar, broadcast per object
        acc = fmaf(w, bf16_to_f32(v_in[(size_t)i * HDIM + tid]), acc);
    }
    emb_out[(size_t)sid * HDIM + tid] = acc;
}

// ---------------------------------------------------------------------------
extern "C" void kernel_launch(void* const* d_in, const int* in_sizes, int n_in,
                              void* d_out, int out_size, void* d_ws, size_t ws_size,
                              hipStream_t stream) {
    const float* objects = (const float*)d_in[0];
    const float* context = (const float*)d_in[1];
    const int*   seg     = (const int*)d_in[2];
    // d_in[3] = num_segments (constant BATCH)
    const float* W0  = (const float*)d_in[4];
    const float* b0  = (const float*)d_in[5];
    const float* W1  = (const float*)d_in[6];
    const float* b1  = (const float*)d_in[7];
    const float* Wkv = (const float*)d_in[8];
    const float* bkv = (const float*)d_in[9];
    const float* Wq  = (const float*)d_in[10];
    const float* bq  = (const float*)d_in[11];

    // workspace partition (256B aligned)
    size_t off = 0;
    auto wsAlloc = [&](size_t bytes) -> void* {
        void* p = (char*)d_ws + off;
        off += (bytes + 255) & ~(size_t)255;
        return p;
    };
    unsigned short* w0b  = (unsigned short*)wsAlloc((size_t)(OSIZE/32) * (HDIM/16)   * 512 * 2);
    unsigned short* w1b  = (unsigned short*)wsAlloc((size_t)(HDIM/32)  * (HDIM/16)   * 512 * 2);
    unsigned short* wkvb = (unsigned short*)wsAlloc((size_t)(HDIM/32)  * (2*HDIM/16) * 512 * 2);
    float*          qws    = (float*)wsAlloc((size_t)BATCH * HDIM * 4);
    float*          logits = (float*)wsAlloc((size_t)N_TOTAL * 4);
    unsigned short* vws    = (unsigned short*)wsAlloc((size_t)N_TOTAL * HDIM * 2);

    // 1) pre-swizzle weights into WMMA B-fragment order (bf16)
    {
        int f0 = (OSIZE/32) * (HDIM/16)   * 32;   // threads
        int f1 = (HDIM/32)  * (HDIM/16)   * 32;
        int f2 = (HDIM/32)  * (2*HDIM/16) * 32;
        pack_weight_frags<<<(f0 + 255)/256, 256, 0, stream>>>(W0,  w0b,  OSIZE, HDIM);
        pack_weight_frags<<<(f1 + 255)/256, 256, 0, stream>>>(W1,  w1b,  HDIM,  HDIM);
        pack_weight_frags<<<(f2 + 255)/256, 256, 0, stream>>>(Wkv, wkvb, HDIM,  2*HDIM);
    }

    // 2) q = context @ Wq + bq
    compute_q<<<BATCH, 256, 0, stream>>>(context, Wq, bq, qws);

    // 3) fused MLP chain + kv + logits
    {
        size_t smem = (size_t)2 * M_TILE * XS * sizeof(unsigned short); // 67584 B
        fused_mlp_attn<<<N_TOTAL / M_TILE, 128, smem, stream>>>(
            objects, seg, w0b, w1b, wkvb, b0, b1, bkv, qws, vws, logits);
    }

    // 4) segment softmax + embedding; outputs concatenated: [emb | w]
    {
        float* emb   = (float*)d_out;
        float* w_out = (float*)d_out + (size_t)BATCH * HDIM;
        segment_softmax_embed<<<BATCH, 256, 0, stream>>>(logits, seg, vws, emb, w_out);
    }
}
